// RNN_23072564314230
// MI455X (gfx1250) — matrix operand order repaired and checked by reading, not compiled, not run on hardware
//
#include <hip/hip_runtime.h>
#include <math.h>

// ---------------------------------------------------------------------------
// Problem constants (from reference): x[B,T,IN] -> 2-layer tanh RNN (H) -> FC(OUT)
// ---------------------------------------------------------------------------
constexpr int Bc   = 64;
constexpr int Tc   = 512;
constexpr int INc  = 128;
constexpr int Hc   = 512;
constexpr int OUTc = 75;
constexpr int OUTP = 80;          // OUT padded to multiple of 16
constexpr int BT   = Bc * Tc;     // 32768 rows

typedef __attribute__((ext_vector_type(16))) __bf16 bf16x16;
typedef __attribute__((ext_vector_type(8)))  __bf16 bf16x8;
typedef __attribute__((ext_vector_type(8)))  float  f32x8;

union ABfrag { bf16x16 v; bf16x8 h[2]; };

// A-fragment (16x32 bf16) per-lane loader from a bf16 row.
// Lane layout (ISA 7.12.2): lanes 0-15 hold K = kb+0..7 (elems 0..7) and
// kb+16..23 (elems 8..15); lanes 16-31 hold kb+8..15 and kb+24..31.
// Caller passes rowk = row + kb + ((lane>=16)?8:0).
__device__ __forceinline__ bf16x16 loadA_bf16(const __bf16* rowk) {
  ABfrag u;
  u.h[0] = *(const bf16x8*)(rowk);
  u.h[1] = *(const bf16x8*)(rowk + 16);
  return u.v;
}

// Same, converting from fp32 on the fly.
__device__ __forceinline__ bf16x16 loadA_f32(const float* rowk) {
  bf16x16 v;
#pragma unroll
  for (int j = 0; j < 8; ++j) {
    v[j]     = (__bf16)rowk[j];
    v[j + 8] = (__bf16)rowk[j + 16];
  }
  return v;
}

// ---------------------------------------------------------------------------
// Weight packing: W[Npad x K] fp32 (row-major, rows >= Nreal zero-padded) ->
// bf16 fragments. Fragment f = nTile*(K/32)+kTile holds 32 lanes x 16 bf16:
// lane l -> n = nTile*16 + l%16, k = kTile*32 + (l/16)*8 + j%8 + (j/8)*16  (j = 0..15),
// i.e. exactly the WMMA B-operand (K x 16) register order.
// ---------------------------------------------------------------------------
__global__ void pack_weights_kernel(const float* __restrict__ W,
                                    __bf16* __restrict__ P,
                                    int Npad, int Nreal, int K) {
  int idx = blockIdx.x * blockDim.x + threadIdx.x;
  if (idx >= Npad * K) return;
  int f     = idx >> 9;        // / 512
  int r     = idx & 511;
  int lane  = r >> 4;          // 0..31
  int j     = r & 15;
  int kTiles = K >> 5;
  int nTile  = f / kTiles;
  int kTile  = f - nTile * kTiles;
  int n = nTile * 16 + (lane & 15);
  int k = kTile * 32 + (lane >> 4) * 8 + (j & 7) + (j >> 3) * 16;
  float v = (n < Nreal) ? W[(size_t)n * K + k] : 0.0f;
  P[idx] = (__bf16)v;
}

// ---------------------------------------------------------------------------
// Layer-0 input projection: xw[t,b,n] = x[b,t,:] . Wih0[n,:] + bih[n] + bhh[n]
// X rows are m = b*T + t (fp32, K=IN); output rows remapped to t*B + b.
// One 16x16 tile per wave; K = 128 -> 4 WMMAs.
// ---------------------------------------------------------------------------
__global__ void proj0_kernel(const float* __restrict__ X,
                             const __bf16* __restrict__ pW,
                             const float* __restrict__ bih,
                             const float* __restrict__ bhh,
                             float* __restrict__ xw) {
  const int lane  = threadIdx.x & 31;
  const int tile  = blockIdx.x * 8 + (threadIdx.x >> 5);
  const int nTiles = Hc / 16;                   // 32
  const int mTile = tile / nTiles;
  const int nTile = tile - mTile * nTiles;
  const int m     = mTile * 16 + (lane & 15);
  const int khalf = (lane >> 4) ? 8 : 0;

  const float* xrow = X + (size_t)m * INc;
  f32x8 acc = {};
#pragma unroll
  for (int kt = 0; kt < INc / 32; ++kt) {
    bf16x16 a = loadA_f32(xrow + kt * 32 + khalf);
    bf16x16 b = *(const bf16x16*)(pW + (((size_t)(nTile * (INc / 32) + kt)) << 9) + lane * 16);
    acc = __builtin_amdgcn_wmma_f32_16x16x32_bf16(false, a, false, b,
                                                  (short)0, acc, false, false);
  }
  const int n     = nTile * 16 + (lane & 15);
  const int mbase = mTile * 16 + ((lane >> 4) ? 8 : 0);
  const float bias = bih[n] + bhh[n];
#pragma unroll
  for (int j = 0; j < 8; ++j) {
    int mm = mbase + j;
    int b  = mm / Tc;               // row -> (b, t); tile never crosses b (16 | T)
    int t  = mm - b * Tc;
    xw[((size_t)t * Bc + b) * Hc + n] = acc[j] + bias;
  }
}

// ---------------------------------------------------------------------------
// Layer-1 input projection: identity row mapping, A is bf16 h0seq[T,B,H], K=H.
// ---------------------------------------------------------------------------
__global__ void proj1_kernel(const __bf16* __restrict__ Hin,
                             const __bf16* __restrict__ pW,
                             const float* __restrict__ bih,
                             const float* __restrict__ bhh,
                             float* __restrict__ xw) {
  const int lane  = threadIdx.x & 31;
  const int tile  = blockIdx.x * 8 + (threadIdx.x >> 5);
  const int nTiles = Hc / 16;
  const int mTile = tile / nTiles;
  const int nTile = tile - mTile * nTiles;
  const int m     = mTile * 16 + (lane & 15);
  const int khalf = (lane >> 4) ? 8 : 0;

  const __bf16* arow = Hin + (size_t)m * Hc;
  f32x8 acc = {};
#pragma unroll
  for (int kt = 0; kt < Hc / 32; ++kt) {
    bf16x16 a = loadA_bf16(arow + kt * 32 + khalf);
    bf16x16 b = *(const bf16x16*)(pW + (((size_t)(nTile * (Hc / 32) + kt)) << 9) + lane * 16);
    acc = __builtin_amdgcn_wmma_f32_16x16x32_bf16(false, a, false, b,
                                                  (short)0, acc, false, false);
  }
  const int n     = nTile * 16 + (lane & 15);
  const int mbase = mTile * 16 + ((lane >> 4) ? 8 : 0);
  const float bias = bih[n] + bhh[n];
#pragma unroll
  for (int j = 0; j < 8; ++j) {
    int mm = mbase + j;
    xw[(size_t)mm * Hc + n] = acc[j] + bias;
  }
}

// ---------------------------------------------------------------------------
// Persistent recurrent pass: h_t = tanh(xw_t + h_{t-1} @ Whh^T), T steps.
// Grid MUST be 16 blocks x 256 threads = 128 waves = all 128 (4x32) 16x16
// tiles of the [64 x 512] state.  Block b owns mTile = b>>2 (16 rows) and
// nTiles (b&3)*8 .. +7 (128 columns).
//
// LDS usage (dynamic, 144 KB):
//   sW [128 KB]: this block's 8 nTile-columns of W_hh, staged ONCE as a
//                single contiguous copy (packed-fragment order), feeding the
//                WMMA B operand from LDS for all 512 steps.
//   sH [ 16 KB]: h_{t-1} rows of this block's mTile, staged once per step
//                via GLOBAL_LOAD_ASYNC_TO_LDS_B128 (ASYNCcnt-tracked DMA,
//                no VGPR round trip).
//
// Steps separated by a device-scope atomic grid barrier (16 blocks are
// trivially co-resident). h stored bf16 [T,B,H]; xw fp32 [T,B,H].
// ---------------------------------------------------------------------------
__global__ void recurrent_kernel(const float* __restrict__ xw,
                                 const __bf16* __restrict__ pWhh,
                                 __bf16* __restrict__ hseq,
                                 unsigned int* __restrict__ ctr) {
  extern __shared__ __bf16 smem[];
  __bf16* sW = smem;                       // 8*16 fragments * 512 = 65536 bf16
  __bf16* sH = smem + 8 * 16 * 512;        // 16 rows * 512 = 8192 bf16

  const int tid   = threadIdx.x;
  const int lane  = tid & 31;
  const int wave  = tid >> 5;                    // 0..7  -> local nTile
  const int nbase = (blockIdx.x & 3) * 8;        // first nTile of this block
  const int mTile = blockIdx.x >> 2;             // 0..3
  const int nTile = nbase + wave;
  const int khalf = (lane >> 4) ? 8 : 0;
  const int n     = nTile * 16 + (lane & 15);
  const int mbase = mTile * 16 + ((lane >> 4) ? 8 : 0);

  // LDS byte offset of sH (low 32 bits of the shared-aperture address).
  const unsigned int sHbase = (unsigned int)(size_t)(void*)sH;

  // ---- Stage this block's W_hh fragments into LDS (contiguous 128 KB) ----
  {
    const bf16x8* src = (const bf16x8*)(pWhh + ((size_t)nbase * 16) * 512);
    bf16x8*       dst = (bf16x8*)sW;
#pragma unroll 4
    for (int i = tid; i < 8 * 16 * 512 / 8; i += 256) dst[i] = src[i];
  }
  __syncthreads();

  for (int t = 0; t < Tc; ++t) {
    // Hoist the xw reads: they only depend on t (precomputed buffer), so
    // they issue before / alongside the staging + WMMA chain.
    float xv[8];
#pragma unroll
    for (int j = 0; j < 8; ++j) {
      xv[j] = xw[((size_t)t * Bc + (mbase + j)) * Hc + n];
    }

    f32x8 acc = {};
    if (t > 0) {
      // ---- Async-stage h_{t-1} rows [mTile*16 .. +15] into LDS (16 KB) ----
      {
        const char* gsrc =
            (const char*)(hseq + ((size_t)(t - 1) * Bc + mTile * 16) * Hc);
#pragma unroll
        for (int i = 0; i < 4; ++i) {
          const int idx = tid + i * 256;                 // 16-byte units
          const unsigned int loff = sHbase + (unsigned int)(idx * 16);
          const void* ga = gsrc + (size_t)idx * 16;
          asm volatile("global_load_async_to_lds_b128 %0, %1, off"
                       :: "v"(loff), "v"(ga)
                       : "memory");
        }
        asm volatile("s_wait_asynccnt 0" ::: "memory");
      }
      __syncthreads();

      const __bf16* hrow = sH + (lane & 15) * Hc;
#pragma unroll
      for (int kt = 0; kt < Hc / 32; ++kt) {
        bf16x16 a = loadA_bf16(hrow + kt * 32 + khalf);
        bf16x16 b = *(const bf16x16*)(sW + ((wave * 16 + kt) << 9) + lane * 16);
        acc = __builtin_amdgcn_wmma_f32_16x16x32_bf16(false, a, false, b,
                                                      (short)0, acc, false, false);
      }
    }
#pragma unroll
    for (int j = 0; j < 8; ++j) {
      int mm = mbase + j;
      size_t off = ((size_t)t * Bc + mm) * Hc + n;
      float hv = tanhf(acc[j] + xv[j]);
      hseq[off] = (__bf16)hv;
    }
    // ---- device-scope grid barrier between time steps ----
    __threadfence();
    __syncthreads();
    if (tid == 0) {
      __hip_atomic_fetch_add(ctr, 1u, __ATOMIC_ACQ_REL, __HIP_MEMORY_SCOPE_AGENT);
      const unsigned int target = (unsigned int)(t + 1) * gridDim.x;
      while (__hip_atomic_load(ctr, __ATOMIC_ACQUIRE, __HIP_MEMORY_SCOPE_AGENT) < target) {
        __builtin_amdgcn_s_sleep(1);
      }
    }
    __syncthreads();
    __threadfence();
  }
}

// ---------------------------------------------------------------------------
// Final FC: out[b,t,o] = h1[t,b,:] . Wfc[o,:] + bfc[o].  K=H, N padded to 80.
// ---------------------------------------------------------------------------
__global__ void fc_kernel(const __bf16* __restrict__ Hin,
                          const __bf16* __restrict__ pW,
                          const float* __restrict__ bfc,
                          float* __restrict__ out) {
  const int lane  = threadIdx.x & 31;
  const int tile  = blockIdx.x * 8 + (threadIdx.x >> 5);
  const int nTiles = OUTP / 16;                 // 5
  const int mTile = tile / nTiles;
  const int nTile = tile - mTile * nTiles;
  const int m     = mTile * 16 + (lane & 15);
  const int khalf = (lane >> 4) ? 8 : 0;

  const __bf16* arow = Hin + (size_t)m * Hc;
  f32x8 acc = {};
#pragma unroll
  for (int kt = 0; kt < Hc / 32; ++kt) {
    bf16x16 a = loadA_bf16(arow + kt * 32 + khalf);
    bf16x16 b = *(const bf16x16*)(pW + (((size_t)(nTile * (Hc / 32) + kt)) << 9) + lane * 16);
    acc = __builtin_amdgcn_wmma_f32_16x16x32_bf16(false, a, false, b,
                                                  (short)0, acc, false, false);
  }
  const int n     = nTile * 16 + (lane & 15);
  const int mbase = mTile * 16 + ((lane >> 4) ? 8 : 0);
  if (n < OUTc) {
    const float bias = bfc[n];
#pragma unroll
    for (int j = 0; j < 8; ++j) {
      int mm = mbase + j;
      int t  = mm / Bc;            // rows are t*B + b; tile never crosses t
      int b  = mm - t * Bc;
      out[((size_t)b * Tc + t) * OUTc + n] = acc[j] + bias;
    }
  }
}

// ---------------------------------------------------------------------------
// Host-side orchestration
// ---------------------------------------------------------------------------
extern "C" void kernel_launch(void* const* d_in, const int* in_sizes, int n_in,
                              void* d_out, int out_size, void* d_ws, size_t ws_size,
                              hipStream_t stream) {
  const float* x    = (const float*)d_in[0];
  const float* Wih0 = (const float*)d_in[1];
  const float* Whh0 = (const float*)d_in[2];
  const float* bih0 = (const float*)d_in[3];
  const float* bhh0 = (const float*)d_in[4];
  const float* Wih1 = (const float*)d_in[5];
  const float* Whh1 = (const float*)d_in[6];
  const float* bih1 = (const float*)d_in[7];
  const float* bhh1 = (const float*)d_in[8];
  const float* Wfc  = (const float*)d_in[9];
  const float* bfc  = (const float*)d_in[10];
  float* out = (float*)d_out;

  char* ws = (char*)d_ws;
  size_t off = 0;
  unsigned int* ctr0 = (unsigned int*)(ws + 0);
  unsigned int* ctr1 = (unsigned int*)(ws + 128);
  off = 256;
  __bf16* pWih0 = (__bf16*)(ws + off); off += (size_t)Hc * INc * 2;
  __bf16* pWhh0 = (__bf16*)(ws + off); off += (size_t)Hc * Hc * 2;
  __bf16* pWih1 = (__bf16*)(ws + off); off += (size_t)Hc * Hc * 2;
  __bf16* pWhh1 = (__bf16*)(ws + off); off += (size_t)Hc * Hc * 2;
  __bf16* pWfc  = (__bf16*)(ws + off); off += (size_t)OUTP * Hc * 2;
  off = (off + 255) & ~(size_t)255;
  float*  xw = (float*)(ws + off);  off += (size_t)Tc * Bc * Hc * 4;  // shared by both layers
  __bf16* h0 = (__bf16*)(ws + off); off += (size_t)Tc * Bc * Hc * 2;
  __bf16* h1 = (__bf16*)(ws + off); off += (size_t)Tc * Bc * Hc * 2;

  // Barrier counters must start at zero every call (ws is poisoned once).
  hipMemsetAsync(ctr0, 0, sizeof(unsigned int), stream);
  hipMemsetAsync(ctr1, 0, sizeof(unsigned int), stream);

  // Pack weights into bf16 WMMA B-fragments.
  {
    int n0 = Hc * INc, n1 = Hc * Hc, n2 = OUTP * Hc;
    pack_weights_kernel<<<(n0 + 255) / 256, 256, 0, stream>>>(Wih0, pWih0, Hc, Hc, INc);
    pack_weights_kernel<<<(n1 + 255) / 256, 256, 0, stream>>>(Whh0, pWhh0, Hc, Hc, Hc);
    pack_weights_kernel<<<(n1 + 255) / 256, 256, 0, stream>>>(Wih1, pWih1, Hc, Hc, Hc);
    pack_weights_kernel<<<(n1 + 255) / 256, 256, 0, stream>>>(Whh1, pWhh1, Hc, Hc, Hc);
    pack_weights_kernel<<<(n2 + 255) / 256, 256, 0, stream>>>(Wfc,  pWfc,  OUTP, OUTc, Hc);
  }

  const int projBlocks = (BT / 16) * (Hc / 16) / 8;    // 8192
  const int fcBlocks   = (BT / 16) * (OUTP / 16) / 8;  // 1280
  const size_t recurLds = (8 * 16 * 512 + 16 * 512) * sizeof(__bf16);  // 144 KB

  // Layer 0
  proj0_kernel<<<projBlocks, 256, 0, stream>>>(x, pWih0, bih0, bhh0, xw);
  recurrent_kernel<<<16, 256, recurLds, stream>>>(xw, pWhh0, h0, ctr0);
  // Layer 1
  proj1_kernel<<<projBlocks, 256, 0, stream>>>(h0, pWih1, bih1, bhh1, xw);
  recurrent_kernel<<<16, 256, recurLds, stream>>>(xw, pWhh1, h1, ctr1);
  // Output projection
  fc_kernel<<<fcBlocks, 256, 0, stream>>>(h1, pWfc, bfc, out);

  (void)in_sizes; (void)n_in; (void)out_size; (void)ws_size;
}
